// cufi_nufft_68032281968976
// MI455X (gfx1250) — matrix operand order repaired and checked by reading, not compiled, not run on hardware
//
#include <hip/hip_runtime.h>

// ---------------------------------------------------------------------------
// NUDFT type-2 on gfx1250 via bf16-split WMMA (fp32-accurate 3-product scheme)
//   ksp[c,k] = sum_x Ex[k,x] * ( sum_y img[c,x,y] * Ey[k,y] )
// Per workgroup: 2 k-subtiles of 16 (B register tiles reused across both ->
// half the L2 traffic). Rolled x/K loops keep VGPRs < 256 (no spills):
//   A = Ey tile, 4 bf16-split planes in LDS, reloaded per K-step
//   B = img^T, y contiguous -> 2x global b128 per lane straight from L2;
//       -img_i operand formed by bf16 sign-flip (v_xor) instead of extra planes
// ---------------------------------------------------------------------------

typedef __attribute__((ext_vector_type(16))) __bf16 v16bf;
typedef __attribute__((ext_vector_type(8)))  __bf16 v8bf;
typedef __attribute__((ext_vector_type(8)))  float  v8f;

#define NC     12
#define NX     96
#define NY     96
#define NK     16384
#define KT     16                 // WMMA M (k rows per subtile)
#define KTILES 2                  // k-subtiles per workgroup
#define KPW    (KT * KTILES)      // 32 k-samples per workgroup
#define WAVES  NC                 // one wave per coil
#define TPB    (WAVES * 32)
#define NTOT   (NC * NX * NY)     // 110592 elements per plane

union V16U { v16bf v; v8bf h[2]; unsigned int u[8]; };

#if __has_builtin(__builtin_amdgcn_cosf) && __has_builtin(__builtin_amdgcn_sinf)
#define FCOS(f) __builtin_amdgcn_cosf(f)   // hardware v_cos_f32: cos(2*pi*f)
#define FSIN(f) __builtin_amdgcn_sinf(f)
#else
#define FCOS(f) __cosf((f) * 6.28318530717958647692f)
#define FSIN(f) __sinf((f) * 6.28318530717958647692f)
#endif

#define WMMA_BF16(A, B, C) \
    __builtin_amdgcn_wmma_f32_16x16x32_bf16(false, (A), false, (B), (short)0, (C), false, false)

// --- prep: split fp32 image into bf16 hi/lo planes (r_hi, r_lo, i_hi, i_lo) --
__global__ void nudft_prep_split(const float* __restrict__ img_r,
                                 const float* __restrict__ img_i,
                                 unsigned short* __restrict__ ws, int n) {
    int t = blockIdx.x * blockDim.x + threadIdx.x;
    if (t >= n) return;
    __bf16* rh = (__bf16*)ws;
    __bf16* rl = rh + NTOT;
    __bf16* ih = rl + NTOT;
    __bf16* il = ih + NTOT;
    float a = img_r[t];
    __bf16 ah = (__bf16)a;
    rh[t] = ah;
    rl[t] = (__bf16)(a - (float)ah);
    float b = img_i[t];
    __bf16 bh = (__bf16)b;
    ih[t] = bh;
    il[t] = (__bf16)(b - (float)bh);
}

// --- main: one workgroup per 32 k-samples; wave w handles coil c = w --------
__launch_bounds__(TPB)
__global__ void nudft_wmma(const unsigned short* __restrict__ ws,
                           const float* __restrict__ trj,
                           float* __restrict__ out) {
    // Ey tables, bf16 split: 0=r_hi 1=r_lo 2=i_hi 3=i_lo
    __shared__ __align__(16) __bf16 sEy[4][KPW][NY];
    __shared__ __align__(16) float2 sEx[KPW][NX];     // (cos, sin) per (k, x)

    const int tid   = threadIdx.x;
    const int ktile = blockIdx.x;

    // ---- build per-tile phase tables: f in TURNS, f = -k_i * r_i ----
    for (int e = tid; e < KPW * NY; e += TPB) {
        int k = e / NY, y = e % NY;
        float ty = trj[(ktile * KPW + k) * 2 + 1];
        float f  = -ty * ((float)(y - NY / 2) * (1.0f / NY));
        float c  = FCOS(f), s = FSIN(f);
        __bf16 ch = (__bf16)c;
        __bf16 sh = (__bf16)s;
        sEy[0][k][y] = ch;  sEy[1][k][y] = (__bf16)(c - (float)ch);
        sEy[2][k][y] = sh;  sEy[3][k][y] = (__bf16)(s - (float)sh);
    }
    for (int e = tid; e < KPW * NX; e += TPB) {
        int k = e / NX, x = e % NX;
        float tx = trj[(ktile * KPW + k) * 2 + 0];
        float f  = -tx * ((float)(x - NX / 2) * (1.0f / NX));
        sEx[k][x] = make_float2(FCOS(f), FSIN(f));
    }
    __syncthreads();

    const __bf16* rh = (const __bf16*)ws;
    const __bf16* rl = rh + NTOT;
    const __bf16* ih = rl + NTOT;
    const __bf16* il = ih + NTOT;

    const int wave = tid >> 5;        // coil index
    const int lane = tid & 31;
    const int half = lane >> 4;       // 0: D rows k=0..7, 1: rows k=8..15
    const int col  = lane & 15;       // A row (k) / B,D column (x) index
    const int c    = wave;

    float kr[KTILES][8], ki[KTILES][8];
#pragma unroll
    for (int kt = 0; kt < KTILES; ++kt)
#pragma unroll
        for (int j = 0; j < 8; ++j) { kr[kt][j] = 0.0f; ki[kt][j] = 0.0f; }

#pragma unroll 1
    for (int xt = 0; xt < NX / 16; ++xt) {
        const int x = xt * 16 + col;
        v8f accR[KTILES], accI[KTILES];   // T_r, T_i chains per k-subtile
#pragma unroll
        for (int kt = 0; kt < KTILES; ++kt) {
            accR[kt] = (v8f){0,0,0,0,0,0,0,0};
            accI[kt] = (v8f){0,0,0,0,0,0,0,0};
        }
#pragma unroll 1
        for (int ks = 0; ks < NY / 32; ++ks) {
            const int y0 = ks * 32;
            // B (16-bit 32x16): lane<16 -> K runs [y0..+7],[y0+16..+23] of column x;
            //                   lane>=16 -> [y0+8..+15],[y0+24..+31]
            const int yB = y0 + 8 * half;
            const size_t base = ((size_t)c * NX + x) * NY + yB;
            V16U bRh, bRl, bIh, bIl, bNh, bNl;
            bRh.h[0] = *(const v8bf*)(rh + base); bRh.h[1] = *(const v8bf*)(rh + base + 16);
            bRl.h[0] = *(const v8bf*)(rl + base); bRl.h[1] = *(const v8bf*)(rl + base + 16);
            bIh.h[0] = *(const v8bf*)(ih + base); bIh.h[1] = *(const v8bf*)(ih + base + 16);
            bIl.h[0] = *(const v8bf*)(il + base); bIl.h[1] = *(const v8bf*)(il + base + 16);
            // exact bf16 negation of img_i (packed sign-flip), VALU co-issues
#pragma unroll
            for (int q = 0; q < 8; ++q) {
                bNh.u[q] = bIh.u[q] ^ 0x80008000u;
                bNl.u[q] = bIl.u[q] ^ 0x80008000u;
            }
            // A (16-bit 16x32): lane<16 -> K runs [y0..+7],[y0+16..+23];
            //                   lane>=16 -> [y0+8..+15],[y0+24..+31]
            const int yA = y0 + 8 * half;
#pragma unroll
            for (int kt = 0; kt < KTILES; ++kt) {
                const int row = kt * KT + col;
                V16U aRh, aRl, aIh, aIl;
                aRh.h[0] = *(const v8bf*)&sEy[0][row][yA];
                aRh.h[1] = *(const v8bf*)&sEy[0][row][yA + 16];
                aRl.h[0] = *(const v8bf*)&sEy[1][row][yA];
                aRl.h[1] = *(const v8bf*)&sEy[1][row][yA + 16];
                aIh.h[0] = *(const v8bf*)&sEy[2][row][yA];
                aIh.h[1] = *(const v8bf*)&sEy[2][row][yA + 16];
                aIl.h[0] = *(const v8bf*)&sEy[3][row][yA];
                aIl.h[1] = *(const v8bf*)&sEy[3][row][yA + 16];
                // T_r = ey_r*img_r + ey_i*(-img_i)   (single wmma chain)
                accR[kt] = WMMA_BF16(aRh.v, bRh.v, accR[kt]);
                accR[kt] = WMMA_BF16(aRh.v, bRl.v, accR[kt]);
                accR[kt] = WMMA_BF16(aRl.v, bRh.v, accR[kt]);
                accR[kt] = WMMA_BF16(aIh.v, bNh.v, accR[kt]);
                accR[kt] = WMMA_BF16(aIh.v, bNl.v, accR[kt]);
                accR[kt] = WMMA_BF16(aIl.v, bNh.v, accR[kt]);
                // T_i = ey_r*img_i + ey_i*img_r      (single wmma chain)
                accI[kt] = WMMA_BF16(aRh.v, bIh.v, accI[kt]);
                accI[kt] = WMMA_BF16(aRh.v, bIl.v, accI[kt]);
                accI[kt] = WMMA_BF16(aRl.v, bIh.v, accI[kt]);
                accI[kt] = WMMA_BF16(aIh.v, bRh.v, accI[kt]);
                accI[kt] = WMMA_BF16(aIh.v, bRl.v, accI[kt]);
                accI[kt] = WMMA_BF16(aIl.v, bRh.v, accI[kt]);
            }
        }
        // Epilogue: ksp += T * Ex[k,x]   (D-tile row k = j + 8*half)
#pragma unroll
        for (int kt = 0; kt < KTILES; ++kt) {
#pragma unroll
            for (int j = 0; j < 8; ++j) {
                float tr = accR[kt][j];
                float ti = accI[kt][j];
                float2 e = sEx[kt * KT + j + 8 * half][x];
                kr[kt][j] += tr * e.x - ti * e.y;
                ki[kt][j] += tr * e.y + ti * e.x;
            }
        }
    }

    // Sum over the 16 x-columns held by each 16-lane group (xor stays in-group)
#pragma unroll
    for (int m = 1; m < 16; m <<= 1) {
#pragma unroll
        for (int kt = 0; kt < KTILES; ++kt)
#pragma unroll
            for (int j = 0; j < 8; ++j) {
                kr[kt][j] += __shfl_xor(kr[kt][j], m, 32);
                ki[kt][j] += __shfl_xor(ki[kt][j], m, 32);
            }
    }

    if (col == 0) {
#pragma unroll
        for (int kt = 0; kt < KTILES; ++kt)
#pragma unroll
            for (int j = 0; j < 8; ++j) {
                int k = ktile * KPW + kt * KT + j + 8 * half;
                *(float2*)(out + ((size_t)c * NK + k) * 2) = make_float2(kr[kt][j], ki[kt][j]);
            }
    }
}

extern "C" void kernel_launch(void* const* d_in, const int* in_sizes, int n_in,
                              void* d_out, int out_size, void* d_ws, size_t ws_size,
                              hipStream_t stream) {
    const float* img_r = (const float*)d_in[0];
    const float* img_i = (const float*)d_in[1];
    const float* trj   = (const float*)d_in[2];
    float* out = (float*)d_out;
    unsigned short* ws = (unsigned short*)d_ws;   // 4 bf16 planes: 884,736 bytes

    nudft_prep_split<<<(NTOT + 255) / 256, 256, 0, stream>>>(img_r, img_i, ws, NTOT);
    nudft_wmma<<<NK / KPW, TPB, 0, stream>>>(ws, trj, out);
}
